// AttentionHead_35768487641415
// MI455X (gfx1250) — matrix operand order repaired and checked by reading, not compiled, not run on hardware
//
#include <hip/hip_runtime.h>

typedef __attribute__((ext_vector_type(16))) _Float16 v16h;
typedef __attribute__((ext_vector_type(8)))  _Float16 v8h;
typedef __attribute__((ext_vector_type(8)))  float    v8f;
typedef __attribute__((ext_vector_type(4)))  float    v4f;

#define EMBED 1024
#define HEAD  64
#define BB    8
#define TT    4096

static __device__ __forceinline__ v8f wmma_f16(v16h a, v16h b, v8f c) {
  // D = A(16x32 f16) * B(32x16 f16) + C(16x16 f32)
  return __builtin_amdgcn_wmma_f32_16x16x32_f16(false, a, false, b, (short)0, c,
                                                false, false);
}
static __device__ __forceinline__ v16h cat8(v8h lo, v8h hi) {
  return __builtin_shufflevector(lo, hi, 0, 1, 2, 3, 4, 5, 6, 7,
                                         8, 9, 10, 11, 12, 13, 14, 15);
}

// ---------------------------------------------------------------------------
// Kernel 0: Wt[p][h][k] = (f16) W_p[k][h]   (B-operand friendly layout)
// ---------------------------------------------------------------------------
__global__ void __launch_bounds__(256)
wconv_kernel(const float* __restrict__ Wk, const float* __restrict__ Wq,
             const float* __restrict__ Wv, _Float16* __restrict__ Wt) {
  int idx = blockIdx.x * 256 + threadIdx.x;         // 3*64*1024 total
  int p   = idx / (HEAD * EMBED);
  int rem = idx - p * (HEAD * EMBED);
  int h   = rem / EMBED;
  int k   = rem - h * EMBED;
  const float* W = (p == 0) ? Wk : ((p == 1) ? Wq : Wv);
  Wt[idx] = (_Float16)W[k * HEAD + h];
}

// ---------------------------------------------------------------------------
// Kernel 1: K/Q/V projections, software-pipelined: B tile i+1 issued before
// the WMMA consuming tile i; next A chunk issued before the chunk's last WMMA.
// K-proj pre-scaled by 1/sqrt(HEAD). V written transposed: Vt[b][h][t].
// ---------------------------------------------------------------------------
__global__ void __launch_bounds__(256)
proj_kernel(const float* __restrict__ x, const _Float16* __restrict__ Wt,
            _Float16* __restrict__ Kp, _Float16* __restrict__ Qp,
            _Float16* __restrict__ Vt) {
  const int lane = threadIdx.x & 31;
  const int wave = threadIdx.x >> 5;
  const int tile = blockIdx.x * 8 + wave;           // 0..2047
  const size_t row0 = (size_t)tile * 16;
  const int b  = tile >> 8;                         // 256 row-tiles per batch
  const int t0 = (tile & 255) << 4;
  const int n  = lane & 15;
  const int hi = lane >> 4;

  v8f acc[3][4];
#pragma unroll
  for (int p = 0; p < 3; ++p)
#pragma unroll
    for (int ct = 0; ct < 4; ++ct) acc[p][ct] = (v8f){};

  const float* arow = x + (row0 + n) * EMBED;       // A row m = lane&15

  // preload raw A chunk 0 (A-layout: a[0..15]=row[kb+16hi..+16))
  v4f g0 = *(const v4f*)(arow + 16 * hi);
  v4f g1 = *(const v4f*)(arow + 16 * hi + 4);
  v4f g2 = *(const v4f*)(arow + 16 * hi + 8);
  v4f g3 = *(const v4f*)(arow + 16 * hi + 12);

  for (int c = 0; c < EMBED / 32; ++c) {
    v16h a;
#pragma unroll
    for (int j = 0; j < 4; ++j) {
      a[j]      = (_Float16)g0[j];
      a[j + 4]  = (_Float16)g1[j];
      a[j + 8]  = (_Float16)g2[j];
      a[j + 12] = (_Float16)g3[j];
    }
    const int kbn = ((c + 1 < EMBED / 32) ? c + 1 : c) * 32 + 16 * hi;

    // B[k][h]: lane column h = n+16ct, elements j -> k = 32c + 16hi + j
    v16h bcur = *(const v16h*)(Wt + ((size_t)n) * EMBED + c * 32 + 16 * hi);
#pragma unroll
    for (int i = 0; i < 12; ++i) {
      const int p = i >> 2, ct = i & 3;
      v16h bnext = bcur;
      if (i < 11) {
        const int pn = (i + 1) >> 2, ctn = (i + 1) & 3;
        bnext = *(const v16h*)(Wt + ((size_t)pn * HEAD + n + 16 * ctn) * EMBED +
                               c * 32 + 16 * hi);
      } else {
        // issue next A chunk before the final WMMA of this chunk
        g0 = *(const v4f*)(arow + kbn);
        g1 = *(const v4f*)(arow + kbn + 4);
        g2 = *(const v4f*)(arow + kbn + 8);
        g3 = *(const v4f*)(arow + kbn + 12);
      }
      acc[p][ct] = wmma_f16(a, bcur, acc[p][ct]);
      bcur = bnext;
    }
  }

#pragma unroll
  for (int ct = 0; ct < 4; ++ct) {
    const int h = n + 16 * ct;
#pragma unroll
    for (int r = 0; r < 8; ++r) {
      const size_t row = row0 + r + 8 * hi;         // C row M = r + 8*hi
      Kp[row * HEAD + h] = (_Float16)(acc[0][ct][r] * 0.125f); // fold 1/sqrt(64)
      Qp[row * HEAD + h] = (_Float16)acc[1][ct][r];
      Vt[((size_t)b * HEAD + h) * TT + t0 + r + 8 * hi] = (_Float16)acc[2][ct][r];
    }
  }
}

// ---------------------------------------------------------------------------
// Kernel 2: causal flash attention (scores = k q^T as in the reference:
// query rows use the K-projection, softmax keys use the Q-projection).
// Key/value tiles are staged into double-buffered LDS with CDNA5 async
// copies (global_load_async_to_lds_b128 / s_wait_asynccnt), one tile ahead.
// Score tiles are computed TRANSPOSED so softmax stats are per-lane and the
// exp'd tile is already in A-operand layout for the P*V WMMA.
// ---------------------------------------------------------------------------
__global__ void __launch_bounds__(256)
attn_kernel(const _Float16* __restrict__ Kp, const _Float16* __restrict__ Qp,
            const _Float16* __restrict__ Vt, float* __restrict__ out) {
  // padded rows: 72 (=36 dwords) and 40 (=20 dwords) -> conflict-free b128 reads
  __shared__ __align__(16) _Float16 sK[2][32][72];  // [buf][key][h]
  __shared__ __align__(16) _Float16 sV[2][64][40];  // [buf][h][key]

  const int tid  = threadIdx.x;
  const int lane = tid & 31;
  const int wave = tid >> 5;
  const int tile = blockIdx.x * 8 + wave;           // 0..2047
  const int b  = tile >> 8;
  const int r0 = (tile & 255) << 4;                 // first query row in batch
  const int n  = lane & 15;
  const int hi = lane >> 4;

  const int myNT  = (r0 >> 5) + 1;                  // this wave's causal tiles
  const int ntmax = ((((blockIdx.x * 8 + 7) & 255) << 4) >> 5) + 1;

  // cooperative async copy: one b128 per thread for K (32x64) and V (64x32)
  const int krow = tid >> 3, kch = tid & 7;
  const int vrow = tid >> 2, vch = tid & 3;
  const _Float16* gK = Qp + ((size_t)b * TT + krow) * HEAD + kch * 8;
  const _Float16* gV = Vt + ((size_t)b * HEAD + vrow) * TT + vch * 8;
  const unsigned lK0 = (unsigned)(size_t)&sK[0][krow][kch * 8];
  const unsigned lK1 = (unsigned)(size_t)&sK[1][krow][kch * 8];
  const unsigned lV0 = (unsigned)(size_t)&sV[0][vrow][vch * 8];
  const unsigned lV1 = (unsigned)(size_t)&sV[1][vrow][vch * 8];

  auto issue_tile = [&](int buf, int s0) {
    unsigned long long aK = (unsigned long long)(size_t)(gK + (size_t)s0 * HEAD);
    unsigned long long aV = (unsigned long long)(size_t)(gV + s0);
    unsigned dK = buf ? lK1 : lK0;
    unsigned dV = buf ? lV1 : lV0;
    asm volatile("global_load_async_to_lds_b128 %0, %1, off"
                 :: "v"(dK), "v"(aK) : "memory");
    asm volatile("global_load_async_to_lds_b128 %0, %1, off"
                 :: "v"(dV), "v"(aV) : "memory");
  };

  // Loop-invariant B operand: Qr^T (64 x 16), column n = query row r0+n.
  const _Float16* qbase = Kp + ((size_t)b * TT + r0 + n) * HEAD;
  const v16h Bq0 = cat8(*(const v8h*)(qbase + 8 * hi), *(const v8h*)(qbase + 16 + 8 * hi));
  const v16h Bq1 = cat8(*(const v8h*)(qbase + 32 + 8 * hi), *(const v8h*)(qbase + 48 + 8 * hi));

  v8f o0 = {}, o1 = {}, o2 = {}, o3 = {};           // O[m=r+8hi][h=n+16ct]
  float Mcol = -3.0e30f, Lcol = 0.0f;               // per-query stats (col n)

  issue_tile(0, 0);                                  // prologue: tile 0

  for (int st = 0; st < ntmax; ++st) {
    const int buf = st & 1;
    if (st + 1 < ntmax) {                            // prefetch next tile
      issue_tile(buf ^ 1, (st + 1) << 5);
      asm volatile("s_wait_asynccnt 0x2" ::: "memory"); // tile st's 2 ops done
    } else {
      asm volatile("s_wait_asynccnt 0x0" ::: "memory");
    }
    __syncthreads();                                 // LDS tile visible to all

    if (st < myNT) {                                 // wave-uniform: EXEC all-1s
      const int s0 = st << 5;

      // --- S^T tiles from LDS: A = key rows, B = Bq0/Bq1 ---
      v8f sA = {}, sB = {};
      {
        const _Float16* kr = &sK[buf][n][0];
        v16h a0 = cat8(*(const v8h*)(kr + 8 * hi),
                       *(const v8h*)(kr + 8 * hi + 16));
        v16h a1 = cat8(*(const v8h*)(kr + 32 + 8 * hi),
                       *(const v8h*)(kr + 48 + 8 * hi));
        sA = wmma_f16(a0, Bq0, sA);
        sA = wmma_f16(a1, Bq1, sA);
      }
      {
        const _Float16* kr = &sK[buf][16 + n][0];
        v16h a0 = cat8(*(const v8h*)(kr + 8 * hi),
                       *(const v8h*)(kr + 8 * hi + 16));
        v16h a1 = cat8(*(const v8h*)(kr + 32 + 8 * hi),
                       *(const v8h*)(kr + 48 + 8 * hi));
        sB = wmma_f16(a0, Bq0, sB);
        sB = wmma_f16(a1, Bq1, sB);
      }

      // --- causal mask: only this wave's last tile crosses the diagonal ---
      if (st == myNT - 1) {
        const int t = r0 + n;                        // this lane's query column
#pragma unroll
        for (int r = 0; r < 8; ++r) {
          if (s0 + r + 8 * hi > t)      sA[r] = -3.0e30f;
          if (s0 + 16 + r + 8 * hi > t) sB[r] = -3.0e30f;
        }
      }

      // --- online softmax (stats per query column, held per-lane) ---
      float tmax = -3.0e30f;
#pragma unroll
      for (int r = 0; r < 8; ++r) tmax = fmaxf(tmax, fmaxf(sA[r], sB[r]));
      tmax = fmaxf(tmax, __shfl_xor(tmax, 16, 32));
      const float Mnew = fmaxf(Mcol, tmax);
      const float cs   = __expf(Mcol - Mnew);
      Mcol = Mnew;

      // exp'd tile converts in-lane into the A-layout of the P*V GEMM
      v16h Pa;
      float tsum = 0.0f;
#pragma unroll
      for (int r = 0; r < 8; ++r) {
        float e0 = __expf(sA[r] - Mnew);
        float e1 = __expf(sB[r] - Mnew);
        tsum += e0 + e1;
        Pa[r]     = (_Float16)e0;
        Pa[r + 8] = (_Float16)e1;
      }
      tsum += __shfl_xor(tsum, 16, 32);
      Lcol = Lcol * cs + tsum;

      // rescale O rows: row m = r+8hi needs the stat of query column m
#pragma unroll
      for (int r = 0; r < 8; ++r) {
        float f = __shfl(cs, 8 * hi + r, 32);
        o0[r] *= f; o1[r] *= f; o2[r] *= f; o3[r] *= f;
      }

      // --- O += P (16x32) * V (32x64); B operand from LDS V^T rows ---
#pragma unroll
      for (int ct = 0; ct < 4; ++ct) {
        const _Float16* vr = &sV[buf][n + 16 * ct][0];
        v16h bv = cat8(*(const v8h*)(vr + 8 * hi),
                       *(const v8h*)(vr + 8 * hi + 16));
        switch (ct) {
          case 0: o0 = wmma_f16(Pa, bv, o0); break;
          case 1: o1 = wmma_f16(Pa, bv, o1); break;
          case 2: o2 = wmma_f16(Pa, bv, o2); break;
          case 3: o3 = wmma_f16(Pa, bv, o3); break;
        }
      }
    }
    __syncthreads();              // protect buf before it is re-filled
  }

  // --- epilogue: divide each row by its softmax sum, write fp32 ---
#pragma unroll
  for (int r = 0; r < 8; ++r) {
    float Lr  = __shfl(Lcol, 8 * hi + r, 32);
    float inv = 1.0f / Lr;
    float* orow = out + ((size_t)b * TT + r0 + r + 8 * hi) * HEAD + n;
    orow[0]  = o0[r] * inv;
    orow[16] = o1[r] * inv;
    orow[32] = o2[r] * inv;
    orow[48] = o3[r] * inv;
  }
}

// ---------------------------------------------------------------------------
extern "C" void kernel_launch(void* const* d_in, const int* in_sizes, int n_in,
                              void* d_out, int out_size, void* d_ws,
                              size_t ws_size, hipStream_t stream) {
  (void)in_sizes; (void)n_in; (void)out_size; (void)ws_size;
  const float* x  = (const float*)d_in[0];
  const float* Wk = (const float*)d_in[1];
  const float* Wq = (const float*)d_in[2];
  const float* Wv = (const float*)d_in[3];
  float* out = (float*)d_out;

  // Workspace layout (f16): Kp | Qp | Vt | Wt  = 4MB + 4MB + 4MB + 384KB
  _Float16* Kp = (_Float16*)d_ws;
  _Float16* Qp = Kp + (size_t)BB * TT * HEAD;
  _Float16* Vt = Qp + (size_t)BB * TT * HEAD;
  _Float16* Wt = Vt + (size_t)BB * TT * HEAD;

  wconv_kernel<<<dim3(3 * HEAD * EMBED / 256), dim3(256), 0, stream>>>(Wk, Wq, Wv, Wt);
  proj_kernel<<<dim3(BB * TT / 16 / 8), dim3(256), 0, stream>>>(x, Wt, Kp, Qp, Vt);
  attn_kernel<<<dim3(BB * TT / 16 / 8), dim3(256), 0, stream>>>(Kp, Qp, Vt, out);
}
